// DiffPoolEncoder_59605556134268
// MI455X (gfx1250) — matrix operand order repaired and checked by reading, not compiled, not run on hardware
//
#include <hip/hip_runtime.h>
#include <hip/hip_bf16.h>

typedef __attribute__((ext_vector_type(16))) _Float16 v16h;
typedef __attribute__((ext_vector_type(8)))  _Float16 v8h;
typedef __attribute__((ext_vector_type(8)))  float    v8f;
typedef __attribute__((ext_vector_type(8)))  float    v8fv;

#define NODES   100
#define NGRAPH  32
#define HEADS   4
#define HID     64
#define DH      256          // HEADS*HID
#define MAXP    100
#define MAXP16  112          // roundup16(MAXP)
#define MAXK32  128          // roundup32(MAXP), fixed LDS stride

// ---------------------------------------------------------------------------
// Weight prep: W (K x N, fp32 row-major) -> Wt (N16 x K, f16), rows n>=N zero.
// Gives the WMMA B-fragment a contiguous 32-byte read per lane per k-step.
// ---------------------------------------------------------------------------
__global__ void prep_weight(const float* __restrict__ W, _Float16* __restrict__ Wt,
                            int K, int N, int N16)
{
    int idx = blockIdx.x * blockDim.x + threadIdx.x;
    if (idx >= N16 * K) return;
    int n = idx / K, k = idx % K;
    Wt[idx] = (n < N) ? (_Float16)W[(size_t)k * N + n] : (_Float16)0.f;
}

// ---------------------------------------------------------------------------
// WMMA GEMM: C[M,N] = act(A[M,K] @ B[K,N] + bias), A fp32 row-major,
// B pre-transposed f16 (N16 x K, zero-padded). K % 32 == 0, M % 16 == 0.
// One wave per 16x16 tile; EXEC all-ones as WMMA requires.
// ---------------------------------------------------------------------------
__global__ __launch_bounds__(32)
void wmma_gemm(const float* __restrict__ A, const _Float16* __restrict__ Bt,
               const float* __restrict__ bias, float* __restrict__ C,
               int M, int N, int K, int act)
{
    const int tileM = blockIdx.x * 16;
    const int tileN = blockIdx.y * 16;
    const int lane  = threadIdx.x;
    const int half  = lane >> 4;     // 0 or 1
    const int idx   = lane & 15;

    const float*    arow = A  + (size_t)(tileM + idx) * K;
    const _Float16* brow = Bt + (size_t)(tileN + idx) * K;

    v8f acc = {};
#pragma unroll 2
    for (int k0 = 0; k0 < K; k0 += 32) {
        const int aBase = k0 + half * 16;   // A: lanes16-31 hold K 16..31
        const int bBase = k0 + half * 16;   // B: lanes16-31 hold K 16..31

        v8fv aLo = *(const v8fv*)(arow + aBase);
        v8fv aHi = *(const v8fv*)(arow + aBase + 8);
        v16h a;
#pragma unroll
        for (int e = 0; e < 8; ++e) {
            a[e]     = (_Float16)aLo[e];
            a[e + 8] = (_Float16)aHi[e];
        }
        v16h b = *(const v16h*)(brow + bBase);

        acc = __builtin_amdgcn_wmma_f32_16x16x32_f16(
                  false, a, false, b, (short)0, acc, false, false);
    }
    const int n = tileN + idx;
    if (n < N) {
        float bb = bias ? bias[n] : 0.f;
#pragma unroll
        for (int r = 0; r < 8; ++r) {
            int m = tileM + r + half * 8;
            float v = acc[r] + bb;
            if (act == 1) v = v > 0.f ? v : 0.f;       // relu
            C[(size_t)m * N + n] = v;
        }
    }
}

// ---------------------------------------------------------------------------
// GAT attention layer (one block per (graph, head), 8 waves).
// hw: (n, 256) = (n, HEADS, HID). out: same shape, elu applied.
// Ab: (B, p, p) per-graph adjacency block. mask = Ab>0 || i==j.
// Aggregation att @ hw runs on WMMA from LDS-staged f16 operands.
// ---------------------------------------------------------------------------
__global__ __launch_bounds__(256)
void gat_attention(const float* __restrict__ hw, const float* __restrict__ Ab,
                   const float* __restrict__ a_s, const float* __restrict__ a_d,
                   float* __restrict__ out, int p)
{
    __shared__ float    s_sc[MAXP];
    __shared__ float    d_sc[MAXP];
    __shared__ _Float16 Wh[MAXP16 * MAXK32];   // normalized att weights (A frag)
    __shared__ _Float16 Vt[HID * MAXK32];      // hw transposed (B frag)

    const int g = blockIdx.x / HEADS;
    const int h = blockIdx.x % HEADS;
    const int t = threadIdx.x, nt = blockDim.x;
    const int K32 = (p + 31) & ~31;
    const int P16 = (p + 15) & ~15;
    const float* hwg = hw + (size_t)g * p * DH + h * HID;
    const float* Abg = Ab + (size_t)g * p * p;
    const float* as  = a_s + h * HID;
    const float* ad  = a_d + h * HID;

    // phase 1: per-node source/dest scores
    for (int i = t; i < p; i += nt) {
        const float* row = hwg + (size_t)i * DH;
        float ss = 0.f, dd = 0.f;
        for (int d = 0; d < HID; ++d) { ss += row[d] * as[d]; dd += row[d] * ad[d]; }
        s_sc[i] = ss; d_sc[i] = dd;
    }
    __syncthreads();

    // phase 2: masked row softmax -> normalized f16 weights in Wh (K-padded 0)
    for (int i = t; i < p; i += nt) {
        _Float16* wrow = &Wh[i * MAXK32];
        float m = -1e30f;
        for (int j = 0; j < p; ++j) {
            float e = s_sc[i] + d_sc[j];
            e = e > 0.f ? e : 0.2f * e;                  // leaky_relu(0.2)
            bool ok = (Abg[(size_t)i * p + j] > 0.f) || (i == j);
            if (!ok) e = -300.f;                         // exp underflows to 0
            wrow[j] = (_Float16)e;
            m = fmaxf(m, e);
        }
        float sum = 0.f;
        for (int j = 0; j < p; ++j) {
            float w = __expf((float)wrow[j] - m);
            wrow[j] = (_Float16)w; sum += w;
        }
        float inv = 1.f / sum;
        for (int j = 0; j < p; ++j) wrow[j] = (_Float16)((float)wrow[j] * inv);
        for (int j = p; j < K32; ++j) wrow[j] = (_Float16)0.f;   // zero K pad
    }
    // phase 2b: stage V transposed: Vt[d][j] = hw[j][h*HID+d], K-padded 0
    for (int od = t; od < HID * K32; od += nt) {
        int d = od / K32, j = od % K32;
        Vt[d * MAXK32 + j] = (j < p) ? (_Float16)hwg[(size_t)j * DH + d]
                                     : (_Float16)0.f;
    }
    __syncthreads();

    // phase 3: O = elu(W @ V) via WMMA; one 16x16 tile per wave per step
    const int lane  = t & 31, wv = t >> 5;
    const int halfw = lane >> 4, idx = lane & 15;
    const int tilesN = HID / 16;                 // 4
    const int nTiles = (P16 >> 4) * tilesN;
    for (int tile = wv; tile < nTiles; tile += (nt >> 5)) {
        const int tm = tile / tilesN, tn = tile % tilesN;
        const _Float16* arow = &Wh[(tm * 16 + idx) * MAXK32];
        const _Float16* brow = &Vt[(tn * 16 + idx) * MAXK32];
        v8f acc = {};
        for (int k0 = 0; k0 < K32; k0 += 32) {
            const int aBase = k0 + halfw * 16;
            const int bBase = k0 + halfw * 16;
            v8h alo = *(const v8h*)(arow + aBase);
            v8h ahi = *(const v8h*)(arow + aBase + 8);
            v16h a;
#pragma unroll
            for (int e = 0; e < 8; ++e) { a[e] = alo[e]; a[e + 8] = ahi[e]; }
            v16h b = *(const v16h*)(brow + bBase);
            acc = __builtin_amdgcn_wmma_f32_16x16x32_f16(
                      false, a, false, b, (short)0, acc, false, false);
        }
#pragma unroll
        for (int r = 0; r < 8; ++r) {
            int i = tm * 16 + r + halfw * 8;
            if (i < p) {
                float v = acc[r];
                out[((size_t)g * p + i) * DH + h * HID + (tn * 16 + idx)] =
                    v > 0.f ? v : (__expf(v) - 1.f);     // elu
            }
        }
    }
}

// ---------------------------------------------------------------------------
// Small utility kernels
// ---------------------------------------------------------------------------
__global__ void fill_zero(float* p, int n)
{
    int i = blockIdx.x * blockDim.x + threadIdx.x;
    if (i < n) p[i] = 0.f;
}

__global__ void build_adj(const int* __restrict__ ei, float* __restrict__ Ab, int E)
{
    int e = blockIdx.x * blockDim.x + threadIdx.x;
    if (e >= E) return;
    int src = ei[e], dst = ei[E + e];
    int g = src / NODES, i = src % NODES, j = dst % NODES;
    Ab[((size_t)g * NODES + i) * NODES + j] = 1.0f;   // duplicate 1.0 writes benign
}

__global__ void row_softmax(float* __restrict__ X, int R, int s)
{
    int r = blockIdx.x * blockDim.x + threadIdx.x;
    if (r >= R) return;
    float* row = X + (size_t)r * s;
    float m = -1e30f;
    for (int j = 0; j < s; ++j) m = fmaxf(m, row[j]);
    float sum = 0.f;
    for (int j = 0; j < s; ++j) { float w = __expf(row[j] - m); row[j] = w; sum += w; }
    float inv = 1.f / sum;
    for (int j = 0; j < s; ++j) row[j] *= inv;
}

// Xn[b,s,d] = sum_p S[b,p,s] * Z[b,p,d]
__global__ void pool_x(const float* __restrict__ S, const float* __restrict__ Z,
                       float* __restrict__ Xn, int p, int s, int d)
{
    int idx = blockIdx.x * blockDim.x + threadIdx.x;
    int total = NGRAPH * s * d;
    if (idx >= total) return;
    int b = idx / (s * d), rem = idx % (s * d), si = rem / d, di = rem % d;
    float acc = 0.f;
    for (int pi = 0; pi < p; ++pi)
        acc += S[((size_t)b * p + pi) * s + si] * Z[((size_t)b * p + pi) * d + di];
    Xn[idx] = acc;
}

// T[b,p,t] = sum_q Ab[b,p,q] * S[b,q,t]
__global__ void pool_a1(const float* __restrict__ Ab, const float* __restrict__ S,
                        float* __restrict__ T, int p, int s)
{
    int idx = blockIdx.x * blockDim.x + threadIdx.x;
    int total = NGRAPH * p * s;
    if (idx >= total) return;
    int b = idx / (p * s), rem = idx % (p * s), pi = rem / s, ti = rem % s;
    float acc = 0.f;
    for (int q = 0; q < p; ++q)
        acc += Ab[((size_t)b * p + pi) * p + q] * S[((size_t)b * p + q) * s + ti];
    T[idx] = acc;
}

// An[b,s,t] = sum_p S[b,p,s] * T[b,p,t]
__global__ void pool_a2(const float* __restrict__ S, const float* __restrict__ T,
                        float* __restrict__ An, int p, int s)
{
    int idx = blockIdx.x * blockDim.x + threadIdx.x;
    int total = NGRAPH * s * s;
    if (idx >= total) return;
    int b = idx / (s * s), rem = idx % (s * s), si = rem / s, ti = rem % s;
    float acc = 0.f;
    for (int pi = 0; pi < p; ++pi)
        acc += S[((size_t)b * p + pi) * s + si] * T[((size_t)b * p + pi) * s + ti];
    An[idx] = acc;
}

__global__ void mean_pool(const float* __restrict__ X, float* __restrict__ out,
                          int p, int d)
{
    int idx = blockIdx.x * blockDim.x + threadIdx.x;
    if (idx >= NGRAPH * d) return;
    int b = idx / d, di = idx % d;
    float acc = 0.f;
    for (int i = 0; i < p; ++i) acc += X[((size_t)b * p + i) * d + di];
    out[idx] = acc / (float)p;
}

// ---------------------------------------------------------------------------
// Host orchestration
// ---------------------------------------------------------------------------
static inline int roundup16(int v) { return (v + 15) & ~15; }

static void prep(hipStream_t s, const float* W, _Float16* Wt, int K, int N)
{
    int N16 = roundup16(N);
    int tot = N16 * K;
    prep_weight<<<(tot + 255) / 256, 256, 0, s>>>(W, Wt, K, N, N16);
}

struct EncWt { _Float16 *w1, *w2, *wm1, *wm2; };

static void run_encoder(hipStream_t stream, const float* X, int n, int di,
                        const float* Ab, int p, const float* const* prm,
                        const EncWt& ew, int d_out,
                        float* tA, float* tB, float* out)
{
    // prm leaf order: [0]=gat0.W [1]=gat0.a_s [2]=gat0.a_d [3]=gat1.W
    // [4]=gat1.a_s [5]=gat1.a_d [6]=mlp0.W [7]=mlp0.b [8]=mlp1.W [9]=mlp1.b
    dim3 gDH(n / 16, DH / 16);
    wmma_gemm<<<gDH, 32, 0, stream>>>(X,  ew.w1,  nullptr, tA, n, DH, di, 0);
    gat_attention<<<NGRAPH * HEADS, 256, 0, stream>>>(tA, Ab, prm[1], prm[2], tB, p);
    wmma_gemm<<<gDH, 32, 0, stream>>>(tB, ew.w2,  nullptr, tA, n, DH, DH, 0);
    gat_attention<<<NGRAPH * HEADS, 256, 0, stream>>>(tA, Ab, prm[4], prm[5], tB, p);
    wmma_gemm<<<gDH, 32, 0, stream>>>(tB, ew.wm1, prm[7], tA, n, DH, DH, 1);   // relu
    dim3 gOut(n / 16, roundup16(d_out) / 16);
    wmma_gemm<<<gOut, 32, 0, stream>>>(tA, ew.wm2, prm[9], out, n, d_out, DH, 0);
}

extern "C" void kernel_launch(void* const* d_in, const int* in_sizes, int n_in,
                              void* d_out, int out_size, void* d_ws, size_t ws_size,
                              hipStream_t stream)
{
    const int N = NGRAPH * NODES;            // 3200
    const int E = in_sizes[1] / 2;           // 51200

    const float* x  = (const float*)d_in[0];
    const int*   ei = (const int*)d_in[1];
    // d_in[2] = batch (unused)

    // params pytree (sorted dict keys): embed[0], embed[1], lin1_b, lin1_w,
    // lin2_b, lin2_w, pool[0], pool[1]; each encoder = 10 leaves.
    const float* embed0[10]; const float* embed1[10];
    const float* pool0[10];  const float* pool1[10];
    for (int k = 0; k < 10; ++k) {
        embed0[k] = (const float*)d_in[3 + k];
        embed1[k] = (const float*)d_in[13 + k];
        pool0[k]  = (const float*)d_in[27 + k];
        pool1[k]  = (const float*)d_in[37 + k];
    }
    const float* lin1_b = (const float*)d_in[23];
    const float* lin1_w = (const float*)d_in[24];
    const float* lin2_b = (const float*)d_in[25];
    const float* lin2_w = (const float*)d_in[26];

    // workspace layout: fp32 arena then f16 arena
    float* w = (float*)d_ws;
    float* Ab0 = w;  w += NGRAPH * NODES * NODES;   // 320000
    float* Ab1 = w;  w += NGRAPH * 25 * 25;         // 20000
    float* tA  = w;  w += (size_t)N * DH;           // 819200
    float* tB  = w;  w += (size_t)N * DH;           // 819200
    float* Z0  = w;  w += (size_t)N * 64;           // 204800
    float* S0  = w;  w += (size_t)N * 25;           // 80000
    float* T0  = w;  w += NGRAPH * 100 * 25;        // 80000
    float* X1  = w;  w += NGRAPH * 25 * 64;         // 51200
    float* Z1  = w;  w += NGRAPH * 25 * 64;         // 51200
    float* S1  = w;  w += NGRAPH * 25 * 5;          // 4000
    float* X2  = w;  w += NGRAPH * 5 * 64;          // 10240
    float* pooled = w; w += NGRAPH * 64;            // 2048
    float* hlin   = w; w += NGRAPH * DH;            // 8192

    _Float16* hw16 = (_Float16*)w;                  // f16 arena (32B-aligned)
    auto halloc = [&](int n16, int K) { _Float16* p = hw16; hw16 += (size_t)n16 * K; return p; };

    EncWt ewE0{ halloc(DH, 64), halloc(DH, DH), halloc(DH, DH), halloc(64, DH) };
    EncWt ewE1{ halloc(DH, 64), halloc(DH, DH), halloc(DH, DH), halloc(64, DH) };
    EncWt ewP0{ halloc(DH, 64), halloc(DH, DH), halloc(DH, DH), halloc(32, DH) };
    EncWt ewP1{ halloc(DH, 64), halloc(DH, DH), halloc(DH, DH), halloc(16, DH) };
    _Float16* lin1_t = halloc(DH, 64);
    _Float16* lin2_t = halloc(128, DH);

    // 0) pre-transpose + f16-convert all weights (zero-padded ragged N)
    const float* const* encs[4] = { embed0, embed1, pool0, pool1 };
    EncWt* ews[4] = { &ewE0, &ewE1, &ewP0, &ewP1 };
    int douts[4]  = { 64, 64, 25, 5 };
    for (int e = 0; e < 4; ++e) {
        prep(stream, encs[e][0], ews[e]->w1,  64, DH);
        prep(stream, encs[e][3], ews[e]->w2,  DH, DH);
        prep(stream, encs[e][6], ews[e]->wm1, DH, DH);
        prep(stream, encs[e][8], ews[e]->wm2, DH, douts[e]);
    }
    prep(stream, lin1_w, lin1_t, 64, DH);
    prep(stream, lin2_w, lin2_t, DH, 128);

    // 1) per-graph adjacency blocks (A is exactly block-diagonal)
    int nAb0 = NGRAPH * NODES * NODES;
    fill_zero<<<(nAb0 + 255) / 256, 256, 0, stream>>>(Ab0, nAb0);
    build_adj<<<(E + 255) / 256, 256, 0, stream>>>(ei, Ab0, E);

    // 2) level 0: pool + embed encoders (n=3200, p=100, d_in=64)
    run_encoder(stream, x, N, 64, Ab0, NODES, pool0,  ewP0, 25, tA, tB, S0);
    run_encoder(stream, x, N, 64, Ab0, NODES, embed0, ewE0, 64, tA, tB, Z0);
    row_softmax<<<(N + 255) / 256, 256, 0, stream>>>(S0, N, 25);
    {
        int tx = NGRAPH * 25 * 64;
        pool_x<<<(tx + 255) / 256, 256, 0, stream>>>(S0, Z0, X1, 100, 25, 64);
        int t1 = NGRAPH * 100 * 25;
        pool_a1<<<(t1 + 255) / 256, 256, 0, stream>>>(Ab0, S0, T0, 100, 25);
        int t2 = NGRAPH * 25 * 25;
        pool_a2<<<(t2 + 255) / 256, 256, 0, stream>>>(S0, T0, Ab1, 100, 25);
    }

    // 3) level 1: n=800, p=25, d_in=64
    const int N1 = NGRAPH * 25;
    run_encoder(stream, X1, N1, 64, Ab1, 25, pool1,  ewP1, 5,  tA, tB, S1);
    run_encoder(stream, X1, N1, 64, Ab1, 25, embed1, ewE1, 64, tA, tB, Z1);
    row_softmax<<<(N1 + 255) / 256, 256, 0, stream>>>(S1, N1, 5);
    {
        int tx = NGRAPH * 5 * 64;
        pool_x<<<(tx + 255) / 256, 256, 0, stream>>>(S1, Z1, X2, 25, 5, 64);
        // An at the last level is computed by the reference but never used.
    }

    // 4) readout head
    int tm = NGRAPH * 64;
    mean_pool<<<(tm + 255) / 256, 256, 0, stream>>>(X2, pooled, 5, 64);
    dim3 gL1(NGRAPH / 16, DH / 16);
    wmma_gemm<<<gL1, 32, 0, stream>>>(pooled, lin1_t, lin1_b, hlin, NGRAPH, DH, 64, 1);
    dim3 gL2(NGRAPH / 16, 128 / 16);
    wmma_gemm<<<gL2, 32, 0, stream>>>(hlin, lin2_t, lin2_b, (float*)d_out,
                                      NGRAPH, 128, DH, 0);
}
